// MoeLayer_10393820856865
// MI455X (gfx1250) — matrix operand order repaired and checked by reading, not compiled, not run on hardware
//
#include <hip/hip_runtime.h>
#include <cstdint>
#include <cstddef>

// ---- problem constants (T, D, H, E, K=2) ----
static constexpr int TT = 8192;   // tokens
static constexpr int DD = 2048;   // model dim
static constexpr int HH = 7168;   // hidden dim
static constexpr int EE = 8;      // experts

typedef __bf16 bf16_t;
typedef __bf16 v16bf __attribute__((ext_vector_type(16)));
typedef __bf16 v8bf  __attribute__((ext_vector_type(8)));
typedef __bf16 v4bf  __attribute__((ext_vector_type(4)));
typedef float  v8f   __attribute__((ext_vector_type(8)));
typedef int    v4i   __attribute__((ext_vector_type(4)));

// ---------------- async global->LDS copy (guarded) ----------------
#if __has_builtin(__builtin_amdgcn_global_load_async_to_lds_b128) && \
    __has_builtin(__builtin_amdgcn_s_wait_asynccnt)
#define USE_ASYNC_LDS 1
#else
#define USE_ASYNC_LDS 0
#endif

__device__ __forceinline__ void cp16_g2s(bf16_t* lds_dst, const bf16_t* gsrc) {
#if USE_ASYNC_LDS
  // Builtin signature (from diagnostic): (global int4* src, shared int4* dst, imm off, imm cpol)
  __builtin_amdgcn_global_load_async_to_lds_b128(
      (__attribute__((address_space(1))) v4i*)const_cast<bf16_t*>(gsrc),
      (__attribute__((address_space(3))) v4i*)lds_dst, 0, 0);
#else
  *(v8bf*)lds_dst = *(const v8bf*)gsrc;
#endif
}

__device__ __forceinline__ void g2s_wait() {
#if USE_ASYNC_LDS
  __builtin_amdgcn_s_wait_asynccnt(0);
#endif
}

// ---------------- small utility kernels ----------------
__global__ void zero_f32(float* __restrict__ p, long n) {
  long i = (long)blockIdx.x * blockDim.x + threadIdx.x;
  long st = (long)gridDim.x * blockDim.x;
  for (; i < n; i += st) p[i] = 0.0f;
}

__global__ void zero_i32(int* __restrict__ p, int n) {
  int i = blockIdx.x * blockDim.x + threadIdx.x;
  if (i < n) p[i] = 0;
}

__global__ void cvt_f32_bf16_v4(const float* __restrict__ s, bf16_t* __restrict__ d, long n4) {
  long i = (long)blockIdx.x * blockDim.x + threadIdx.x;
  long st = (long)gridDim.x * blockDim.x;
  for (; i < n4; i += st) {
    float4 v = ((const float4*)s)[i];
    v4bf o = { (__bf16)v.x, (__bf16)v.y, (__bf16)v.z, (__bf16)v.w };
    ((v4bf*)d)[i] = o;
  }
}

// w2[E][H][D] (fp32) -> w2t[E][D][H] (bf16), so GEMM2's B-columns are contiguous over K=h
__global__ void cvt_w2_transpose(const float* __restrict__ w2, bf16_t* __restrict__ w2t) {
  const long n = (long)EE * HH * DD;
  long i = (long)blockIdx.x * blockDim.x + threadIdx.x;
  long st = (long)gridDim.x * blockDim.x;
  for (; i < n; i += st) {
    long e = i / ((long)HH * DD);
    long r = i - e * (long)HH * DD;
    long h = r / DD;
    long dcol = r - h * DD;
    w2t[e * (long)DD * HH + dcol * HH + h] = (bf16_t)w2[i];
  }
}

// ---------------- gating: top-2 + softmax + per-expert token lists ----------------
__global__ void moe_gate(const float* __restrict__ x, const float* __restrict__ gw,
                         int* __restrict__ cnt, int* __restrict__ ids,
                         float* __restrict__ wts) {
  const int t = blockIdx.x * 8 + (threadIdx.x >> 5);  // one wave32 per token
  const int lane = threadIdx.x & 31;
  if (t >= TT) return;
  const float* xr = x + (size_t)t * DD;
  float acc[EE];
#pragma unroll
  for (int e = 0; e < EE; ++e) acc[e] = 0.0f;
  for (int d = lane; d < DD; d += 32) {
    float xv = xr[d];
#pragma unroll
    for (int e = 0; e < EE; ++e) acc[e] += xv * gw[e * DD + d];
  }
#pragma unroll
  for (int e = 0; e < EE; ++e) {
#pragma unroll
    for (int off = 16; off > 0; off >>= 1) acc[e] += __shfl_xor(acc[e], off, 32);
  }
  if (lane == 0) {
    int e0 = 0; float v0 = acc[0];
#pragma unroll
    for (int e = 1; e < EE; ++e) if (acc[e] > v0) { v0 = acc[e]; e0 = e; }
    int e1 = (e0 == 0) ? 1 : 0; float v1 = acc[e1];
#pragma unroll
    for (int e = 0; e < EE; ++e) if (e != e0 && acc[e] > v1) { v1 = acc[e]; e1 = e; }
    float wa = 1.0f / (1.0f + __expf(v1 - v0));  // softmax over {v0,v1}
    float wb = 1.0f - wa;
    int p0 = atomicAdd(&cnt[e0], 1);
    ids[e0 * TT + p0] = t; wts[e0 * TT + p0] = wa;
    int p1 = atomicAdd(&cnt[e1], 1);
    ids[e1 * TT + p1] = t; wts[e1 * TT + p1] = wb;
  }
}

// ---------------- GEMM tiling config ----------------
#define BM 64
#define BN 128
#define BK 64
// 256 threads = 8 waves arranged 2 (M) x 4 (N); wave tile 32x32 = 2x2 WMMA tiles.

// Stage 1: G = silu(x_e @ w1e^T) * (x_e @ w3e^T), gathered rows, bf16 output
__global__ __launch_bounds__(256, 1) void moe_gemm13_swiglu(
    const bf16_t* __restrict__ xb, const bf16_t* __restrict__ w1b,
    const bf16_t* __restrict__ w3b, const int* __restrict__ cnt,
    const int* __restrict__ ids, bf16_t* __restrict__ G, int e) {
  const int n_e = cnt[e];
  const int tileM = blockIdx.y;
  if (tileM * BM >= n_e) return;
  const int h0 = blockIdx.x * BN;

  __shared__ alignas(64) bf16_t At[2][BM][BK];
  __shared__ alignas(64) bf16_t B1t[2][BN][BK];
  __shared__ alignas(64) bf16_t B3t[2][BN][BK];
  __shared__ int rowTok[BM];

  const int tid = threadIdx.x;
  if (tid < BM) {
    int grow = tileM * BM + tid;
    rowTok[tid] = (grow < n_e) ? ids[e * TT + grow] : -1;
  }
  __syncthreads();

  const bf16_t* w1p = w1b + (size_t)e * HH * DD;
  const bf16_t* w3p = w3b + (size_t)e * HH * DD;

  auto stage = [&](int k0, int buf) {
#pragma unroll
    for (int i = 0; i < 2; ++i) {  // A tile: 64x64 bf16 = 512 16B chunks
      int c = tid + i * 256;
      int r = c >> 3, cc = (c & 7) * 8;
      int tok = rowTok[r];
      if (tok >= 0) {
        cp16_g2s(&At[buf][r][cc], xb + (size_t)tok * DD + k0 + cc);
      } else {
        v8bf z = {};
        *(v8bf*)&At[buf][r][cc] = z;
      }
    }
#pragma unroll
    for (int i = 0; i < 4; ++i) {  // B tiles: 128x64 each
      int c = tid + i * 256;
      int r = c >> 3, cc = (c & 7) * 8;
      cp16_g2s(&B1t[buf][r][cc], w1p + (size_t)(h0 + r) * DD + k0 + cc);
      cp16_g2s(&B3t[buf][r][cc], w3p + (size_t)(h0 + r) * DD + k0 + cc);
    }
  };

  const int lane = tid & 31;
  const int wid = tid >> 5;
  const int waveM = wid >> 2;  // 0..1
  const int waveN = wid & 3;   // 0..3
  const int half = lane >> 4;  // 0/1
  const int ln = lane & 15;

  v8f acc1[2][2] = {};
  v8f acc3[2][2] = {};

  stage(0, 0);
  g2s_wait();
  __syncthreads();

  int buf = 0;
  for (int k0 = 0; k0 < DD; k0 += BK, buf ^= 1) {
    if (k0 + BK < DD) stage(k0 + BK, buf ^ 1);
#pragma unroll
    for (int kk = 0; kk < BK; kk += 32) {
      v16bf af[2], b1f[2], b3f[2];
#pragma unroll
      for (int mi = 0; mi < 2; ++mi) {
        // A 16x32 bf16 layout: lanes 0-15 K=0..15, lanes 16-31 K=16..31 (contiguous)
        const bf16_t* rp = &At[buf][waveM * 32 + mi * 16 + ln][kk + half * 16];
        v8bf lo = *(const v8bf*)rp;
        v8bf hi = *(const v8bf*)(rp + 8);
        af[mi] = __builtin_shufflevector(lo, hi, 0, 1, 2, 3, 4, 5, 6, 7,
                                         8, 9, 10, 11, 12, 13, 14, 15);
      }
#pragma unroll
      for (int ni = 0; ni < 2; ++ni) {
        // B 32x16: lane = column N; lanes 0-15 K=0..15, lanes 16-31 K=16..31 (contiguous)
        b1f[ni] = *(const v16bf*)&B1t[buf][waveN * 32 + ni * 16 + ln][kk + half * 16];
        b3f[ni] = *(const v16bf*)&B3t[buf][waveN * 32 + ni * 16 + ln][kk + half * 16];
      }
#pragma unroll
      for (int mi = 0; mi < 2; ++mi)
#pragma unroll
        for (int ni = 0; ni < 2; ++ni) {
          acc1[mi][ni] = __builtin_amdgcn_wmma_f32_16x16x32_bf16(
              false, af[mi], false, b1f[ni], (short)0, acc1[mi][ni], false, false);
          acc3[mi][ni] = __builtin_amdgcn_wmma_f32_16x16x32_bf16(
              false, af[mi], false, b3f[ni], (short)0, acc3[mi][ni], false, false);
        }
    }
    g2s_wait();
    __syncthreads();
  }

  // SwiGLU epilogue -> bf16 G (compacted rows)
#pragma unroll
  for (int mi = 0; mi < 2; ++mi)
#pragma unroll
    for (int ni = 0; ni < 2; ++ni) {
      int col = h0 + waveN * 32 + ni * 16 + ln;
#pragma unroll
      for (int r = 0; r < 8; ++r) {
        int rl = waveM * 32 + mi * 16 + r + half * 8;
        int grow = tileM * BM + rl;
        if (grow < n_e) {
          float a = acc1[mi][ni][r];
          float b = acc3[mi][ni][r];
          float g = (a / (1.0f + __expf(-a))) * b;
          G[(size_t)grow * HH + col] = (bf16_t)g;
        }
      }
    }
}

// Stage 2: out[tok] += w_tok * (G @ w2t_e), contraction over H
__global__ __launch_bounds__(256, 1) void moe_gemm2(
    const bf16_t* __restrict__ G, const bf16_t* __restrict__ w2t,
    const int* __restrict__ cnt, const int* __restrict__ ids,
    const float* __restrict__ wts, float* __restrict__ out, int e) {
  const int n_e = cnt[e];
  const int tileM = blockIdx.y;
  if (tileM * BM >= n_e) return;
  const int d0 = blockIdx.x * BN;

  __shared__ alignas(64) bf16_t At[2][BM][BK];
  __shared__ alignas(64) bf16_t Bt[2][BN][BK];
  __shared__ int rowTok[BM];
  __shared__ float rowW[BM];

  const int tid = threadIdx.x;
  if (tid < BM) {
    int grow = tileM * BM + tid;
    if (grow < n_e) {
      rowTok[tid] = ids[e * TT + grow];
      rowW[tid] = wts[e * TT + grow];
    } else {
      rowTok[tid] = -1;
      rowW[tid] = 0.0f;
    }
  }
  __syncthreads();

  const bf16_t* w2p = w2t + (size_t)e * DD * HH;

  auto stage = [&](int k0, int buf) {
#pragma unroll
    for (int i = 0; i < 2; ++i) {
      int c = tid + i * 256;
      int r = c >> 3, cc = (c & 7) * 8;
      if (rowTok[r] >= 0) {
        cp16_g2s(&At[buf][r][cc], G + (size_t)(tileM * BM + r) * HH + k0 + cc);
      } else {
        v8bf z = {};
        *(v8bf*)&At[buf][r][cc] = z;
      }
    }
#pragma unroll
    for (int i = 0; i < 4; ++i) {
      int c = tid + i * 256;
      int r = c >> 3, cc = (c & 7) * 8;
      cp16_g2s(&Bt[buf][r][cc], w2p + (size_t)(d0 + r) * HH + k0 + cc);
    }
  };

  const int lane = tid & 31;
  const int wid = tid >> 5;
  const int waveM = wid >> 2;
  const int waveN = wid & 3;
  const int half = lane >> 4;
  const int ln = lane & 15;

  v8f acc[2][2] = {};

  stage(0, 0);
  g2s_wait();
  __syncthreads();

  int buf = 0;
  for (int k0 = 0; k0 < HH; k0 += BK, buf ^= 1) {
    if (k0 + BK < HH) stage(k0 + BK, buf ^ 1);
#pragma unroll
    for (int kk = 0; kk < BK; kk += 32) {
      v16bf af[2], bf[2];
#pragma unroll
      for (int mi = 0; mi < 2; ++mi) {
        const bf16_t* rp = &At[buf][waveM * 32 + mi * 16 + ln][kk + half * 16];
        v8bf lo = *(const v8bf*)rp;
        v8bf hi = *(const v8bf*)(rp + 8);
        af[mi] = __builtin_shufflevector(lo, hi, 0, 1, 2, 3, 4, 5, 6, 7,
                                         8, 9, 10, 11, 12, 13, 14, 15);
      }
#pragma unroll
      for (int ni = 0; ni < 2; ++ni)
        bf[ni] = *(const v16bf*)&Bt[buf][waveN * 32 + ni * 16 + ln][kk + half * 16];
#pragma unroll
      for (int mi = 0; mi < 2; ++mi)
#pragma unroll
        for (int ni = 0; ni < 2; ++ni)
          acc[mi][ni] = __builtin_amdgcn_wmma_f32_16x16x32_bf16(
              false, af[mi], false, bf[ni], (short)0, acc[mi][ni], false, false);
    }
    g2s_wait();
    __syncthreads();
  }

  // Weighted scatter-accumulate into out. Each (token,col) is touched by exactly
  // one lane within this launch; experts are sequential launches -> no atomics.
#pragma unroll
  for (int mi = 0; mi < 2; ++mi)
#pragma unroll
    for (int ni = 0; ni < 2; ++ni) {
      int col = d0 + waveN * 32 + ni * 16 + ln;
#pragma unroll
      for (int r = 0; r < 8; ++r) {
        int rl = waveM * 32 + mi * 16 + r + half * 8;
        int tok = rowTok[rl];
        if (tok >= 0) {
          size_t o = (size_t)tok * DD + col;
          out[o] += rowW[rl] * acc[mi][ni][r];
        }
      }
    }
}

// ---------------- host launcher ----------------
extern "C" void kernel_launch(void* const* d_in, const int* in_sizes, int n_in,
                              void* d_out, int out_size, void* d_ws, size_t ws_size,
                              hipStream_t stream) {
  (void)in_sizes; (void)n_in; (void)out_size;
  const float* x  = (const float*)d_in[0];
  const float* gw = (const float*)d_in[1];
  const float* w1 = (const float*)d_in[2];
  const float* w2 = (const float*)d_in[3];
  const float* w3 = (const float*)d_in[4];
  float* out = (float*)d_out;

  char* ws = (char*)d_ws;
  size_t off = 0;
  auto take = [&](size_t bytes) -> void* {
    void* p = ws + off;
    off = (off + bytes + 255) & ~(size_t)255;
    return p;
  };
  bf16_t* xb  = (bf16_t*)take((size_t)TT * DD * sizeof(bf16_t));
  bf16_t* w1b = (bf16_t*)take((size_t)EE * HH * DD * sizeof(bf16_t));
  bf16_t* w3b = (bf16_t*)take((size_t)EE * HH * DD * sizeof(bf16_t));
  bf16_t* w2t = (bf16_t*)take((size_t)EE * HH * DD * sizeof(bf16_t));
  bf16_t* G   = (bf16_t*)take((size_t)TT * HH * sizeof(bf16_t));
  int*    cnt = (int*)take(EE * sizeof(int));
  int*    ids = (int*)take((size_t)EE * TT * sizeof(int));
  float*  wts = (float*)take((size_t)EE * TT * sizeof(float));
  if (off > ws_size) return;  // workspace too small: safe no-op

  zero_f32<<<2048, 256, 0, stream>>>(out, (long)TT * DD);
  zero_i32<<<1, 64, 0, stream>>>(cnt, EE);

  cvt_f32_bf16_v4<<<4096, 256, 0, stream>>>(x, xb, (long)TT * DD / 4);
  cvt_f32_bf16_v4<<<8192, 256, 0, stream>>>(w1, w1b, (long)EE * HH * DD / 4);
  cvt_f32_bf16_v4<<<8192, 256, 0, stream>>>(w3, w3b, (long)EE * HH * DD / 4);
  cvt_w2_transpose<<<8192, 256, 0, stream>>>(w2, w2t);

  moe_gate<<<TT / 8, 256, 0, stream>>>(x, gw, cnt, ids, wts);

  for (int e = 0; e < EE; ++e) {
    moe_gemm13_swiglu<<<dim3(HH / BN, TT / BM), 256, 0, stream>>>(
        xb, w1b, w3b, cnt, ids, G, e);
    moe_gemm2<<<dim3(DD / BN, TT / BM), 256, 0, stream>>>(
        G, w2t, cnt, ids, wts, out, e);
  }
}
